// LookupConv2d_60739427500579
// MI455X (gfx1250) — matrix operand order repaired and checked by reading, not compiled, not run on hardware
//
#include <hip/hip_runtime.h>
#include <stdint.h>

typedef __bf16  v16bf __attribute__((ext_vector_type(16)));
typedef float   v8f   __attribute__((ext_vector_type(8)));
typedef int     v4i   __attribute__((ext_vector_type(4)));
typedef unsigned short u16;

union Frag { v4i i4[2]; v16bf b; };

#define CIN   256
#define COUT  512
#define KDIM  2304      // 9 taps * 256 channels, ordered k = r*256 + c
#define HW    3136      // 56*56
#define WIDTH 56
#define BM    256       // Cout tile per block
#define BN    64        // pixel tile per block
#define BK    32        // K step == 32 consecutive channels of one tap
#define KT    72        // KDIM/BK
#define LDK   40        // padded LDS row (bf16): 80B rows -> conflict-free b128 reads

__device__ __forceinline__ unsigned short f2bf(float f) {
  union { float f; unsigned int u; } c; c.f = f;
  unsigned int u = c.u;
  unsigned int r = u + 0x7FFFu + ((u >> 16) & 1u);   // round-to-nearest-even
  return (unsigned short)(r >> 16);
}

// Async global->LDS 16B copy (per lane), tracked by ASYNCcnt.
__device__ __forceinline__ void async_ld_b128(void* lds, const void* g) {
  unsigned lds_off = (unsigned)(size_t)lds;          // generic LDS addr[31:0] == LDS offset
  unsigned long long ga = (unsigned long long)(size_t)g;
  asm volatile("global_load_async_to_lds_b128 %0, %1, off"
               :: "v"(lds_off), "v"(ga) : "memory");
}

__device__ __forceinline__ void wait_async0() {
#if __has_builtin(__builtin_amdgcn_s_wait_asynccnt)
  __builtin_amdgcn_s_wait_asynccnt(0);
#else
  asm volatile("s_wait_asynccnt 0" ::: "memory");
#endif
}

// ---------------------------------------------------------------------------
// Kernel 1: W[o][k], k = r*256 + c (tap-major K), stored bf16.
// ---------------------------------------------------------------------------
__global__ __launch_bounds__(256) void synth_weights(
    const float* __restrict__ dict, const long long* __restrict__ idx,
    const float* __restrict__ coef, u16* __restrict__ wB) {
  int t = blockIdx.x * 256 + threadIdx.x;
  int o = t / KDIM;
  int k = t - o * KDIM;
  int r = k >> 8;
  int c = k & 255;
  float acc = 0.f;
#pragma unroll
  for (int s = 0; s < 3; ++s) {
    int a = (int)idx[o * 3 + s];
    acc += coef[o * 3 + s] * dict[a * KDIM + c * 9 + r];
  }
  wB[t] = f2bf(acc);
}

// ---------------------------------------------------------------------------
// Kernel 2: implicit-GEMM conv, bf16 WMMA / f32 accum.
// Block: 256 threads (8 waves), tile 256 Cout x 64 px, K stepped by 32,
// double-buffered LDS. Wave w owns rows [w*32, w*32+32) x all 64 px
// = 2 A-frags x 4 B-frags = 8 WMMA per K step.
// A tile staged with global_load_async_to_lds_b128 (ASYNCcnt);
// B tile gathered from f32 x (one tap per K chunk -> scalar decode + 1 branch).
// ---------------------------------------------------------------------------
__global__ __launch_bounds__(256) void conv_wmma(
    const float* __restrict__ x, const u16* __restrict__ wB,
    float* __restrict__ out) {
  __shared__ __align__(16) u16 sA[2][BM][LDK];   // 40 KB
  __shared__ __align__(16) u16 sB[2][BN][LDK];   // 10 KB

  const int tid     = threadIdx.x;
  const int pixBase = blockIdx.x * BN;
  const int mBase   = blockIdx.y * BM;
  const int img     = blockIdx.z;

  const int wave = tid >> 5, lane = tid & 31;
  const int lm = lane & 15, half = lane >> 4;

  // A staging: 256 rows x 4 x 16B chunks = 1024 copies -> 4 per thread.
  const int arow0 = tid >> 2;                // base row 0..63 (+64*i)
  const int ach   = tid & 3;                 // 16B chunk in 64B K-row
  const u16* aSrc0 = wB + (size_t)(mBase + arow0) * KDIM + ach * 8;

  // B staging: 64 consecutive threads share one k-octet; lanes sweep pixels.
  const int bp  = tid & 63;
  const int bk8 = (tid >> 6) * 8;
  const int pix = pixBase + bp;
  const int py  = pix / WIDTH;
  const int px  = pix - py * WIDTH;
  const float* xImg = x + (size_t)img * CIN * HW;

  auto stage = [&](int buf, int kt) {
    const int kk = kt * BK;
    // --- A: async memory->LDS, no VGPR round trip ---
#pragma unroll
    for (int i = 0; i < 4; ++i) {
      async_ld_b128(&sA[buf][arow0 + 64 * i][ach * 8],
                    aSrc0 + (size_t)(64 * i) * KDIM + kk);
    }
    // --- B: one tap per chunk; decode + pad test once ---
    const int r  = kt >> 3;
    const int cb = (kt & 7) * BK;
    const int q  = r / 3;
    const int dy = q - 1, dx = (r - q * 3) - 1;
    const int yy = py + dy, xx = px + dx;
    const bool ok = ((unsigned)yy < WIDTH) & ((unsigned)xx < WIDTH);
    v4i pk = {0, 0, 0, 0};                    // bf16 zeros
    if (ok) {
      const float* src = xImg + (size_t)(cb + bk8) * HW + (yy * WIDTH + xx);
#pragma unroll
      for (int i = 0; i < 4; ++i) {
        float v0 = src[(size_t)(2 * i) * HW];
        float v1 = src[(size_t)(2 * i + 1) * HW];
        pk[i] = (int)((unsigned)f2bf(v0) | ((unsigned)f2bf(v1) << 16));
      }
    }
    *(v4i*)&sB[buf][bp][bk8] = pk;            // one 16B LDS store
  };

  v8f acc[2][4] = {};

  stage(0, 0);
  wait_async0();
  __syncthreads();

  for (int kt = 0; kt < KT; ++kt) {
    const int cur = kt & 1;
    if (kt + 1 < KT) stage(cur ^ 1, kt + 1);  // issue next-tile loads early

    // Fragments per CDNA5 WMMA VGPR layouts (16-bit A 16x32, B 32x16).
    Frag a[2], b[4];
#pragma unroll
    for (int tm = 0; tm < 2; ++tm) {
      const v4i* pa = (const v4i*)&sA[cur][wave * 32 + tm * 16 + lm][0];
      a[tm].i4[0] = pa[half];                 // half0: K0-7   half1: K8-15
      a[tm].i4[1] = pa[half + 2];             // half0: K16-23 half1: K24-31
    }
#pragma unroll
    for (int tn = 0; tn < 4; ++tn) {
      const v4i* pb = (const v4i*)&sB[cur][tn * 16 + lm][0];
      b[tn].i4[0] = pb[half];                 // half0: K0-7   half1: K8-15
      b[tn].i4[1] = pb[half + 2];             // half0: K16-23 half1: K24-31
    }

#pragma unroll
    for (int tm = 0; tm < 2; ++tm)
#pragma unroll
      for (int tn = 0; tn < 4; ++tn)
        acc[tm][tn] = __builtin_amdgcn_wmma_f32_16x16x32_bf16(
            false, a[tm].b, false, b[tn].b, (short)0, acc[tm][tn], false, false);

    wait_async0();                            // drain async A-stage
    __syncthreads();
  }

  // Store: C/D layout -> VGPR v, lanes0-15: M=v,N=lane; lanes16-31: M=v+8.
  float* o0 = out + ((size_t)img * COUT + mBase + wave * 32) * HW + pixBase;
#pragma unroll
  for (int tm = 0; tm < 2; ++tm)
#pragma unroll
    for (int tn = 0; tn < 4; ++tn)
#pragma unroll
      for (int v = 0; v < 8; ++v) {
        int orow = tm * 16 + v + 8 * half;
        o0[(size_t)orow * HW + tn * 16 + lm] = acc[tm][tn][v];
      }
}

// ---------------------------------------------------------------------------
extern "C" void kernel_launch(void* const* d_in, const int* in_sizes, int n_in,
                              void* d_out, int out_size, void* d_ws, size_t ws_size,
                              hipStream_t stream) {
  const float*     x    = (const float*)d_in[0];
  const float*     dict = (const float*)d_in[1];
  const long long* idx  = (const long long*)d_in[2];
  const float*     coef = (const float*)d_in[3];
  float*           out  = (float*)d_out;
  u16*             wB   = (u16*)d_ws;            // 512*2304 bf16 = 2.25 MB

  synth_weights<<<dim3((COUT * KDIM) / 256), dim3(256), 0, stream>>>(
      dict, idx, coef, wB);

  dim3 grid(HW / BN, COUT / BM, 16);             // (49, 2, 16)
  conv_wmma<<<grid, dim3(256), 0, stream>>>(x, wB, out);
}
